// AOEncoder_70222715289874
// MI455X (gfx1250) — compile-verified, matrix operand order checked by reading, not run on hardware
//
#include <hip/hip_runtime.h>
#include <hip/hip_bf16.h>
#include <math.h>

// ---------------------------------------------------------------------------
// AOEncoder forward for MI455X (gfx1250, wave32, WMMA).
//  * BiLSTM gate GEMMs on V_WMMA_F32_16X16X32_BF16 (bf16 in, f32 acc).
//  * Input projection fused into the recurrent step: [x_t ; h] @ [Wih ; Whh]^T.
//  * Gate rows permuted so each wave owns i,f,g,o of its own 32-wide hidden
//    slice -> LSTM cell math stays in accumulator registers.
//  * An SGPR *offset* (not the pointer) is laundered per time step: keeps the
//    weight loads GLOBAL (saddr + 24-bit immediate offsets, no FLAT/DScnt
//    coupling, no 64-bit VALU address chains) while still blocking LICM from
//    hoisting+spilling the 128 B-fragments.
//  * self_attn's softmax is over a size-1 axis => alpha == mask exactly, so
//    Wa1/ba1/wa2 are dead and the 18-GFLOP attention GEMM is skipped.
// ---------------------------------------------------------------------------

typedef __attribute__((ext_vector_type(16))) __bf16 v16bf;
typedef __attribute__((ext_vector_type(8)))  float  v8f;

// Problem constants (match reference)
constexpr int B_   = 8,  K_  = 16, NQ_ = 4, NR_ = 4, LQ_ = 64, LR_ = 64;
constexpr int D_   = 128, HID_ = 256, TOPK_ = 5;
constexpr int H2_  = 2 * HID_;        // 512
constexpr int G4_  = 4 * HID_;        // 1024 gates
constexpr int I2_  = 2 * D_;          // 256 LSTM input dim
constexpr int KC_  = I2_ + HID_;      // 512 fused contraction dim
constexpr int L_   = 64;
constexpr int SQ_  = B_ * NQ_;        // 32 query sequences
constexpr int SR_  = B_ * K_ * NR_;   // 512 review sequences
constexpr int S_   = SQ_ + SR_;       // 544
constexpr int DD_  = H2_ + D_;        // 640

// Output section offsets (flat f32, reference tuple order)
constexpr size_t O_HQ  = 0;
constexpr size_t N_HQ  = (size_t)B_*NQ_*LQ_*H2_;          // 1,048,576
constexpr size_t O_S   = O_HQ  + N_HQ;
constexpr size_t N_S   = (size_t)2*B_*HID_;               // 4096
constexpr size_t O_C   = O_S   + N_S;
constexpr size_t O_QA  = O_C   + N_S;
constexpr size_t N_QA  = (size_t)B_*NQ_*D_;               // 4096
constexpr size_t O_HR  = O_QA  + N_QA;
constexpr size_t N_HR  = (size_t)B_*TOPK_*NR_*LR_*H2_;    // 5,242,880
constexpr size_t O_SR  = O_HR  + N_HR;
constexpr size_t N_SR  = (size_t)B_*TOPK_*NR_*H2_;        // 81,920
constexpr size_t O_RA  = O_SR  + N_SR;
constexpr size_t N_RAo = (size_t)B_*TOPK_*NR_*D_;         // 20,480
constexpr size_t O_RO  = O_RA  + N_RAo;
constexpr size_t O_RAM = O_RO  + N_RAo;
constexpr size_t N_RAM = (size_t)B_*TOPK_*NR_;            // 160
constexpr size_t O_RB  = O_RAM + N_RAM;
constexpr size_t N_RB  = (size_t)B_*TOPK_*LR_;            // 2560
constexpr size_t O_RM  = O_RB  + N_RB;
constexpr size_t O_RT  = O_RM  + N_RB;
constexpr size_t N_RT  = (size_t)B_*TOPK_;                // 40
constexpr size_t O_REV = O_RT  + N_RT;
constexpr size_t O_IDX = O_REV + N_RB;
constexpr size_t O_TS  = O_IDX + N_RT;

// ---------------------------------------------------------------------------
// Prep: fp32 weights -> fused, gate-permuted bf16 [Wih ; Whh].
// Permuted row p = w*128 + tt*16 + nn:
//   gate = tt/2 (i,f,g,o), hidden j = w*32 + (tt&1)*16 + nn
// so wave w's 8 accumulator tiles hold i,f,g,o of hidden slice [32w,32w+32).
// ---------------------------------------------------------------------------
__global__ void prep_w_kernel(const float* __restrict__ Wih_f, const float* __restrict__ Whh_f,
                              const float* __restrict__ Wih_b, const float* __restrict__ Whh_b,
                              __bf16* __restrict__ W_cat) {
  int e = blockIdx.x * blockDim.x + threadIdx.x;
  if (e >= 2 * G4_ * KC_) return;
  int dir = e / (G4_ * KC_);
  int r   = e % (G4_ * KC_);
  int p   = r / KC_, kk = r % KC_;
  int w    = p >> 7;
  int tt   = (p >> 4) & 7;
  int nn   = p & 15;
  int gate = tt >> 1;
  int j    = w * 32 + (tt & 1) * 16 + nn;
  int orow = gate * HID_ + j;
  const float* Wih = dir ? Wih_b : Wih_f;
  const float* Whh = dir ? Whh_b : Whh_f;
  float v = (kk < I2_) ? Wih[orow * I2_ + kk] : Whh[orow * HID_ + (kk - I2_)];
  W_cat[e] = (__bf16)v;
}

// ---------------------------------------------------------------------------
// Prep: gather embeddings into bf16 LSTM inputs x_in[(S),(L),(I2)]
// ---------------------------------------------------------------------------
__global__ void prep_x_kernel(const int* __restrict__ que_batch, const int* __restrict__ que_asp,
                              const int* __restrict__ rev_batch, const int* __restrict__ rev_asp,
                              const float* __restrict__ emb, __bf16* __restrict__ x_in) {
  int e = blockIdx.x * blockDim.x + threadIdx.x;
  if (e >= S_ * L_ * I2_) return;
  int j = e % I2_;
  int t = (e / I2_) % L_;
  int s = e / (I2_ * L_);
  int tok;
  if (s < SQ_) {
    int b = s >> 2, nq = s & 3;
    tok = (j < D_) ? que_batch[b * LQ_ + t] : que_asp[b * NQ_ + nq];
  } else {
    int u = s - SQ_;
    int bk = u >> 2;
    int nr = u & 3;
    tok = (j < D_) ? rev_batch[bk * LR_ + t] : rev_asp[bk * NR_ + nr];
  }
  x_in[e] = (__bf16)emb[(size_t)tok * D_ + (j & (D_ - 1))];
}

// ---------------------------------------------------------------------------
// Prep: aspect/opinion embeddings (fp32) for outputs & scoring
// ---------------------------------------------------------------------------
__global__ void prep_asp_kernel(const int* __restrict__ que_asp, const int* __restrict__ rev_asp,
                                const int* __restrict__ rev_opi, const float* __restrict__ emb,
                                float* __restrict__ q_a_ws, float* __restrict__ r_a_ws,
                                float* __restrict__ r_o_ws, float* __restrict__ out_qa) {
  int e = blockIdx.x * blockDim.x + threadIdx.x;
  const int NQA = B_ * NQ_ * D_;
  const int NRA = B_ * K_ * NR_ * D_;
  if (e < NQA) {
    int s = e / D_, c = e % D_;
    float v = emb[(size_t)que_asp[s] * D_ + c];
    q_a_ws[e] = v;
    out_qa[e] = v;
  } else if (e < NQA + NRA) {
    int e2 = e - NQA;
    int u = e2 / D_, c = e2 % D_;
    r_a_ws[e2] = emb[(size_t)rev_asp[u] * D_ + c];
  } else if (e < NQA + 2 * NRA) {
    int e2 = e - NQA - NRA;
    int u = e2 / D_, c = e2 % D_;
    r_o_ws[e2] = emb[(size_t)rev_opi[u] * D_ + c];
  }
}

// ---------------------------------------------------------------------------
// BiLSTM: one workgroup per (dir, 16-sequence tile). 8 waves, WMMA bf16.
// ---------------------------------------------------------------------------
__global__ __launch_bounds__(256)
void lstm_kernel(const __bf16* __restrict__ x_in,     // (S,L,I2)
                 const __bf16* __restrict__ W_cat,    // (2,G4,KC) permuted
                 const float* __restrict__ bih_f, const float* __restrict__ bhh_f,
                 const float* __restrict__ bih_b, const float* __restrict__ bhh_b,
                 const int* __restrict__ que_lens,    // (B)
                 const int* __restrict__ rev_lens,    // (B*K)
                 float* __restrict__ h_cat,           // (S,L,H2)  [fwd|bwd]
                 float* __restrict__ hN,              // (2,S,HID)
                 float* __restrict__ cN) {            // (2,S,HID)
  const int tile = blockIdx.x % (S_ / 16);
  const int dir  = blockIdx.x / (S_ / 16);
  const int seq0 = tile * 16;
  const int tid  = threadIdx.x;
  const int wave = tid >> 5;
  const int lane = tid & 31;
  const int nn   = lane & 15;   // A-row (M) during loads / N column in D
  const int half = lane >> 4;   // K sub-chunk select / M-half in D

  // One contiguous A-operand tile: [0,I2) = x_t, [I2,KC) = h_{t-1}
  __shared__ __bf16 lds_a[16][KC_];    // 16 KB
  __shared__ int    lds_len[16];

  const __bf16* Wd  = W_cat + (size_t)dir * G4_ * KC_;
  const float*  bih = dir ? bih_b : bih_f;
  const float*  bhh = dir ? bhh_b : bhh_f;

  if (tid < 16) {
    int s = seq0 + tid;
    lds_len[tid] = (s < SQ_) ? que_lens[s / NQ_] : rev_lens[(s - SQ_) / NR_];
  }
  for (int e = tid; e < 16 * HID_; e += 256) lds_a[e >> 8][I2_ + (e & 255)] = (__bf16)0.f;
  __syncthreads();

  // per-lane gate biases: one per owned N-tile (constant over M) -> acc init
  float bias_t[8];
#pragma unroll
  for (int tt = 0; tt < 8; ++tt) {
    const int gate = tt >> 1;
    const int j = wave * 32 + (tt & 1) * 16 + nn;
    bias_t[tt] = bih[gate * HID_ + j] + bhh[gate * HID_ + j];
  }
  int len_r[8];
#pragma unroll
  for (int r = 0; r < 8; ++r) len_r[r] = lds_len[r + 8 * half];

  // loop-invariant, per-lane part of the B-fragment address (hoisted once)
  const int laneoff = (wave * 128 + nn) * KC_ + half * 16;   // elements

  float c_reg[16], h_reg[16];
#pragma unroll
  for (int i = 0; i < 16; ++i) { c_reg[i] = 0.f; h_reg[i] = 0.f; }

  for (int t = 0; t < L_; ++t) {
    // ---- stage x_t tile (time-reversed for the backward direction) ----
    {
      const int m  = tid >> 4;
      const int j0 = (tid & 15) * 16;
      const int n  = lds_len[m];
      int ti = t;
      if (dir && t < n) ti = n - 1 - t;
      *(v16bf*)&lds_a[m][j0] =
          *(const v16bf*)(x_in + ((size_t)(seq0 + m) * L_ + ti) * I2_ + j0);
    }
    __syncthreads();

    // Launder an SGPR offset (not the pointer): loads stay GLOBAL with
    // saddr + immediate offsets, but LICM cannot hoist them out of the
    // t-loop (which would spill 128 fragments to scratch).
    size_t woff = 0;
    asm volatile("" : "+s"(woff));
    const __bf16* Wt = Wd + woff;

    v8f acc[8];
#pragma unroll
    for (int n = 0; n < 8; ++n) {
      const float bv = bias_t[n];
      v8f a = {bv, bv, bv, bv, bv, bv, bv, bv};
      acc[n] = a;
    }

#pragma unroll
    for (int kk = 0; kk < KC_ / 32; ++kk) {
      const v16bf afrag = *(const v16bf*)&lds_a[nn][kk * 32 + half * 16];
#pragma unroll
      for (int n = 0; n < 8; ++n) {
        // byte offsets n*16*KC*2 (16 KB) + kk*64 : 24-bit immediate range
        const __bf16* bsrc = Wt + laneoff + (n * 16 * KC_ + kk * 32);
        if (n == 0) __builtin_prefetch(bsrc + 16 * KC_, 0, 1);  // global_prefetch_b8
        const v16bf bfrag = *(const v16bf*)bsrc;
        acc[n] = __builtin_amdgcn_wmma_f32_16x16x32_bf16(
                     false, afrag, false, bfrag, (short)0, acc[n], false, false);
      }
    }
    __syncthreads();   // all WMMA reads of lds_a done before overwrite

    // ---- in-register LSTM cell.  D layout: lane=(m/8)*16+n, vgpr=m%8 ----
#pragma unroll
    for (int q = 0; q < 2; ++q) {
      const int j = wave * 32 + q * 16 + nn;
#pragma unroll
      for (int r = 0; r < 8; ++r) {
        const int m = r + 8 * half;
        const int n = len_r[r];
        const float mt = (t < n) ? 1.f : 0.f;
        float gi = acc[0 + q][r];
        float gf = acc[2 + q][r];
        float gg = acc[4 + q][r];
        float go = acc[6 + q][r];
        gi = 1.f / (1.f + __expf(-gi));
        gf = 1.f / (1.f + __expf(-gf));
        gg = tanhf(gg);
        go = 1.f / (1.f + __expf(-go));
        const float cn = gf * c_reg[q * 8 + r] + gi * gg;
        const float hn = go * tanhf(cn);
        h_reg[q * 8 + r] = mt * hn + (1.f - mt) * h_reg[q * 8 + r];
        c_reg[q * 8 + r] = mt * cn + (1.f - mt) * c_reg[q * 8 + r];
        lds_a[m][I2_ + j] = (__bf16)h_reg[q * 8 + r];
        int po = t;                       // bwd output flipped back in place
        if (dir && t < n) po = n - 1 - t;
        h_cat[((size_t)(seq0 + m) * L_ + po) * H2_ + dir * HID_ + j] = mt * hn;
      }
    }
    __syncthreads();   // h fully updated before next step's WMMA
  }

  // finals for (s, c) outputs
#pragma unroll
  for (int q = 0; q < 2; ++q) {
    const int j = wave * 32 + q * 16 + nn;
#pragma unroll
    for (int r = 0; r < 8; ++r) {
      const int m = r + 8 * half;
      hN[((size_t)dir * S_ + seq0 + m) * HID_ + j] = h_reg[q * 8 + r];
      cN[((size_t)dir * S_ + seq0 + m) * HID_ + j] = c_reg[q * 8 + r];
    }
  }
}

// ---------------------------------------------------------------------------
// h_q output: que_asp_mask * h_cat (query part)
// ---------------------------------------------------------------------------
__global__ void hq_out_kernel(const float* __restrict__ h_cat, const float* __restrict__ qam,
                              float* __restrict__ out) {
  size_t e = (size_t)blockIdx.x * blockDim.x + threadIdx.x;
  if (e >= N_HQ) return;
  int h = e % H2_;
  int l = (e / H2_) % LQ_;
  int s = e / ((size_t)H2_ * LQ_);   // b*NQ+nq
  out[O_HQ + e] = qam[s] * h_cat[((size_t)s * L_ + l) * H2_ + h];
}

// ---------------------------------------------------------------------------
// self_attn (softmax over size-1 axis == identity): masked sum over L.
// ---------------------------------------------------------------------------
__global__ __launch_bounds__(256)
void attn_kernel(const float* __restrict__ h_cat, const float* __restrict__ que_mask,
                 const float* __restrict__ qam, const float* __restrict__ rev_mask,
                 const float* __restrict__ ram, float* __restrict__ s_q_ws,
                 float* __restrict__ s_r_ws) {
  const int s = blockIdx.x, tid = threadIdx.x;
  float scale; const float* mrow; float* dst;
  if (s < SQ_) {
    int b = s / NQ_;
    scale = qam[s]; mrow = que_mask + b * LQ_; dst = s_q_ws + (size_t)s * H2_;
  } else {
    int u = s - SQ_;
    scale = ram[u]; mrow = rev_mask + (u / NR_) * LR_; dst = s_r_ws + (size_t)u * H2_;
  }
  const float* hsrc = h_cat + (size_t)s * L_ * H2_;
  for (int h = tid; h < H2_; h += 256) {
    float acc = 0.f;
    for (int l = 0; l < L_; ++l) acc += mrow[l] * hsrc[l * H2_ + h];
    dst[h] = scale * acc;
  }
}

// ---------------------------------------------------------------------------
// Per-batch combine: s/c outputs, a_s_q (B,DD), a_s_r (B,K,DD)
// ---------------------------------------------------------------------------
__global__ __launch_bounds__(256)
void combine_kernel(const float* __restrict__ hN, const float* __restrict__ cN,
                    const float* __restrict__ qam, const float* __restrict__ ram,
                    const float* __restrict__ s_q_ws, const float* __restrict__ s_r_ws,
                    const float* __restrict__ q_a_ws, const float* __restrict__ r_a_ws,
                    float* __restrict__ a_s_q, float* __restrict__ a_s_r,
                    float* __restrict__ out_s, float* __restrict__ out_c) {
  const int b = blockIdx.x, tid = threadIdx.x;
  float qs = 0.f;
  for (int nq = 0; nq < NQ_; ++nq) qs += qam[b * NQ_ + nq];

  for (int d = 0; d < 2; ++d) {
    const int h = tid;  // HID == blockDim
    float sa = 0.f, ca = 0.f;
    for (int nq = 0; nq < NQ_; ++nq) {
      const float w = qam[b * NQ_ + nq];
      const int s = b * NQ_ + nq;
      sa += w * hN[((size_t)d * S_ + s) * HID_ + h];
      ca += w * cN[((size_t)d * S_ + s) * HID_ + h];
    }
    out_s[(d * B_ + b) * HID_ + h] = sa / qs;
    out_c[(d * B_ + b) * HID_ + h] = ca / qs;
  }
  for (int e = tid; e < DD_; e += 256) {
    float v = 0.f;
    for (int nq = 0; nq < NQ_; ++nq) {
      const int s = b * NQ_ + nq;
      v += (e < H2_) ? s_q_ws[(size_t)s * H2_ + e] : q_a_ws[(size_t)s * D_ + (e - H2_)];
    }
    a_s_q[b * DD_ + e] = v / qs;
  }
  for (int k = 0; k < K_; ++k) {
    const int bk = b * K_ + k;
    float rs = 0.f;
    for (int nr = 0; nr < NR_; ++nr) rs += ram[bk * NR_ + nr];
    for (int e = tid; e < DD_; e += 256) {
      float v = 0.f;
      for (int nr = 0; nr < NR_; ++nr) {
        const int u = bk * NR_ + nr;
        v += (e < H2_) ? s_r_ws[(size_t)u * H2_ + e] : r_a_ws[(size_t)u * D_ + (e - H2_)];
      }
      a_s_r[(size_t)bk * DD_ + e] = v / rs;
    }
  }
}

// ---------------------------------------------------------------------------
// Bilinear score + top-5 (per batch).  scores = tanh(a_s_q @ Wbil @ a_s_r^T)
// ---------------------------------------------------------------------------
__global__ __launch_bounds__(256)
void score_topk_kernel(const float* __restrict__ a_s_q, const float* __restrict__ a_s_r,
                       const float* __restrict__ Wbil, int* __restrict__ idx_ws,
                       float* __restrict__ out_idx, float* __restrict__ out_ts) {
  const int b = blockIdx.x, tid = threadIdx.x;
  __shared__ float v[DD_];
  __shared__ float sc[K_];
  for (int e = tid; e < DD_; e += 256) {
    float a = 0.f;
    for (int d = 0; d < DD_; ++d) a += a_s_q[b * DD_ + d] * Wbil[(size_t)d * DD_ + e];
    v[e] = a;
  }
  __syncthreads();
  if (tid < K_) {
    float a = 0.f;
    const float* r = a_s_r + (size_t)(b * K_ + tid) * DD_;
    for (int e = 0; e < DD_; ++e) a += v[e] * r[e];
    sc[tid] = tanhf(a);
  }
  __syncthreads();
  if (tid == 0) {
    float scl[K_];
    for (int k = 0; k < K_; ++k) scl[k] = sc[k];
    for (int j = 0; j < TOPK_; ++j) {
      int best = 0; float bv = scl[0];
      for (int k = 1; k < K_; ++k) if (scl[k] > bv) { bv = scl[k]; best = k; }
      idx_ws[b * TOPK_ + j] = best;
      out_idx[b * TOPK_ + j] = (float)best;
      out_ts [b * TOPK_ + j] = bv;
      scl[best] = -1e30f;
    }
  }
}

// ---------------------------------------------------------------------------
// sel(h_r): (B,TOPK,NR,LR,H2) gather with rev_asp_mask scale
// ---------------------------------------------------------------------------
__global__ void sel_big_kernel(const float* __restrict__ h_cat, const float* __restrict__ ram,
                               const int* __restrict__ idx_ws, float* __restrict__ out) {
  size_t e = (size_t)blockIdx.x * blockDim.x + threadIdx.x;
  if (e >= N_HR) return;
  int h = e % H2_;
  size_t r = e / H2_;
  int l = r % LR_;  r /= LR_;
  int nr = r % NR_; r /= NR_;
  int j = r % TOPK_;
  int b = r / TOPK_;
  const int k = idx_ws[b * TOPK_ + j];
  const int u = (b * K_ + k) * NR_ + nr;
  out[O_HR + e] = ram[u] * h_cat[((size_t)(SQ_ + u) * L_ + l) * H2_ + h];
}

// ---------------------------------------------------------------------------
// Small sel outputs: s_r, r_a, r_o, ram, rev_batch, rev_mask, ratings, rev_ext
// ---------------------------------------------------------------------------
__global__ void sel_small_kernel(const float* __restrict__ s_r_ws, const float* __restrict__ r_a_ws,
                                 const float* __restrict__ r_o_ws, const float* __restrict__ ram,
                                 const int* __restrict__ rev_batch, const float* __restrict__ rev_mask,
                                 const float* __restrict__ ratings, const int* __restrict__ rev_ext,
                                 const int* __restrict__ idx_ws, float* __restrict__ out) {
  const int C0 = (int)N_SR, C1 = C0 + (int)N_RAo, C2 = C1 + (int)N_RAo;
  const int C3 = C2 + (int)N_RAM, C4 = C3 + (int)N_RB, C5 = C4 + (int)N_RB;
  const int C6 = C5 + (int)N_RT,  C7 = C6 + (int)N_RB;
  int e = blockIdx.x * blockDim.x + threadIdx.x;
  if (e >= C7) return;
  if (e < C0) {
    int h = e % H2_; int r = e / H2_;
    int nr = r % NR_; r /= NR_;
    int j = r % TOPK_; int b = r / TOPK_;
    int u = (b * K_ + idx_ws[b * TOPK_ + j]) * NR_ + nr;
    out[O_SR + e] = s_r_ws[(size_t)u * H2_ + h];
  } else if (e < C1) {
    int e2 = e - C0;
    int c = e2 % D_; int r = e2 / D_;
    int nr = r % NR_; r /= NR_;
    int j = r % TOPK_; int b = r / TOPK_;
    int u = (b * K_ + idx_ws[b * TOPK_ + j]) * NR_ + nr;
    out[O_RA + e2] = r_a_ws[(size_t)u * D_ + c];
  } else if (e < C2) {
    int e2 = e - C1;
    int c = e2 % D_; int r = e2 / D_;
    int nr = r % NR_; r /= NR_;
    int j = r % TOPK_; int b = r / TOPK_;
    int u = (b * K_ + idx_ws[b * TOPK_ + j]) * NR_ + nr;
    out[O_RO + e2] = r_o_ws[(size_t)u * D_ + c];
  } else if (e < C3) {
    int e2 = e - C2;
    int nr = e2 % NR_; int r = e2 / NR_;
    int j = r % TOPK_; int b = r / TOPK_;
    out[O_RAM + e2] = ram[(b * K_ + idx_ws[b * TOPK_ + j]) * NR_ + nr];
  } else if (e < C4) {
    int e2 = e - C3;
    int l = e2 % LR_; int r = e2 / LR_;
    int j = r % TOPK_; int b = r / TOPK_;
    out[O_RB + e2] = (float)rev_batch[(b * K_ + idx_ws[b * TOPK_ + j]) * LR_ + l];
  } else if (e < C5) {
    int e2 = e - C4;
    int l = e2 % LR_; int r = e2 / LR_;
    int j = r % TOPK_; int b = r / TOPK_;
    out[O_RM + e2] = rev_mask[(b * K_ + idx_ws[b * TOPK_ + j]) * LR_ + l];
  } else if (e < C6) {
    int e2 = e - C5;
    int j = e2 % TOPK_; int b = e2 / TOPK_;
    out[O_RT + e2] = ratings[b * K_ + idx_ws[b * TOPK_ + j]];
  } else {
    int e2 = e - C6;
    int l = e2 % LR_; int r = e2 / LR_;
    int j = r % TOPK_; int b = r / TOPK_;
    out[O_REV + e2] = (float)rev_ext[(b * K_ + idx_ws[b * TOPK_ + j]) * LR_ + l];
  }
}

// ---------------------------------------------------------------------------
extern "C" void kernel_launch(void* const* d_in, const int* in_sizes, int n_in,
                              void* d_out, int out_size, void* d_ws, size_t ws_size,
                              hipStream_t stream) {
  (void)in_sizes; (void)n_in; (void)out_size; (void)ws_size;
  const int*   que_batch = (const int*)  d_in[0];
  const int*   que_lens  = (const int*)  d_in[1];
  const float* que_mask  = (const float*)d_in[2];
  const int*   que_asp   = (const int*)  d_in[3];
  const float* qam       = (const float*)d_in[4];
  const int*   rev_batch = (const int*)  d_in[5];
  const int*   rev_lens  = (const int*)  d_in[6];
  const float* rev_mask  = (const float*)d_in[7];
  const int*   rev_asp   = (const int*)  d_in[8];
  const int*   rev_opi   = (const int*)  d_in[9];
  const float* ram       = (const float*)d_in[10];
  const float* ratings   = (const float*)d_in[11];
  const int*   rev_ext   = (const int*)  d_in[12];
  const float* emb       = (const float*)d_in[13];
  const float* Wih_f     = (const float*)d_in[14];
  const float* Whh_f     = (const float*)d_in[15];
  const float* bih_f     = (const float*)d_in[16];
  const float* bhh_f     = (const float*)d_in[17];
  const float* Wih_b     = (const float*)d_in[18];
  const float* Whh_b     = (const float*)d_in[19];
  const float* bih_b     = (const float*)d_in[20];
  const float* bhh_b     = (const float*)d_in[21];
  // d_in[22..24] = Wa1, ba1, wa2: mathematically dead (softmax over size-1 axis)
  const float* Wbil      = (const float*)d_in[25];
  float* out = (float*)d_out;

  // workspace carve (~95 MB)
  char* p = (char*)d_ws;
  auto take = [&](size_t bytes) { char* q = p; p += (bytes + 255) & ~(size_t)255; return q; };
  __bf16* x_in   = (__bf16*)take(sizeof(__bf16) * (size_t)S_ * L_ * I2_);
  __bf16* W_cat  = (__bf16*)take(sizeof(__bf16) * 2 * G4_ * KC_);
  float*  h_cat  = (float*) take(sizeof(float) * (size_t)S_ * L_ * H2_);
  float*  hN     = (float*) take(sizeof(float) * 2 * S_ * HID_);
  float*  cNw    = (float*) take(sizeof(float) * 2 * S_ * HID_);
  float*  q_a_ws = (float*) take(sizeof(float) * B_ * NQ_ * D_);
  float*  r_a_ws = (float*) take(sizeof(float) * B_ * K_ * NR_ * D_);
  float*  r_o_ws = (float*) take(sizeof(float) * B_ * K_ * NR_ * D_);
  float*  s_q_ws = (float*) take(sizeof(float) * SQ_ * H2_);
  float*  s_r_ws = (float*) take(sizeof(float) * SR_ * H2_);
  float*  a_s_q  = (float*) take(sizeof(float) * B_ * DD_);
  float*  a_s_r  = (float*) take(sizeof(float) * B_ * K_ * DD_);
  int*    idx_ws = (int*)   take(sizeof(int) * B_ * TOPK_);

  const int T = 256;
  prep_w_kernel<<<(2 * G4_ * KC_ + T - 1) / T, T, 0, stream>>>(Wih_f, Whh_f, Wih_b, Whh_b, W_cat);
  prep_x_kernel<<<(S_ * L_ * I2_ + T - 1) / T, T, 0, stream>>>(que_batch, que_asp, rev_batch,
                                                               rev_asp, emb, x_in);
  {
    const int n = B_ * NQ_ * D_ + 2 * B_ * K_ * NR_ * D_;
    prep_asp_kernel<<<(n + T - 1) / T, T, 0, stream>>>(que_asp, rev_asp, rev_opi, emb,
                                                       q_a_ws, r_a_ws, r_o_ws, out + O_QA);
  }
  lstm_kernel<<<2 * (S_ / 16), T, 0, stream>>>(x_in, W_cat, bih_f, bhh_f, bih_b, bhh_b,
                                               que_lens, rev_lens, h_cat, hN, cNw);
  hq_out_kernel<<<(int)((N_HQ + T - 1) / T), T, 0, stream>>>(h_cat, qam, out);
  attn_kernel<<<S_, T, 0, stream>>>(h_cat, que_mask, qam, rev_mask, ram, s_q_ws, s_r_ws);
  combine_kernel<<<B_, T, 0, stream>>>(hN, cNw, qam, ram, s_q_ws, s_r_ws, q_a_ws, r_a_ws,
                                       a_s_q, a_s_r, out + O_S, out + O_C);
  score_topk_kernel<<<B_, T, 0, stream>>>(a_s_q, a_s_r, Wbil, idx_ws, out + O_IDX, out + O_TS);
  sel_big_kernel<<<(int)((N_HR + T - 1) / T), T, 0, stream>>>(h_cat, ram, idx_ws, out);
  {
    const int n = (int)(N_SR + 2 * N_RAo + N_RAM + 3 * N_RB + N_RT);
    sel_small_kernel<<<(n + T - 1) / T, T, 0, stream>>>(s_r_ws, r_a_ws, r_o_ws, ram, rev_batch,
                                                        rev_mask, ratings, rev_ext, idx_ws, out);
  }
}
